// NativeBitInferenceLinear_19799799235325
// MI455X (gfx1250) — matrix operand order repaired and checked by reading, not compiled, not run on hardware
//
#include <hip/hip_runtime.h>

typedef __bf16 bf16;
typedef bf16  v16bf __attribute__((ext_vector_type(16)));
typedef float v8f   __attribute__((ext_vector_type(8)));

#define M_DIM 8192
#define K_DIM 4096
#define N_DIM 4096
#define LDSS  40   // bf16 per LDS row: 32 data + 8 pad -> 80B stride, conflict-free b128 reads

__device__ __forceinline__ bf16 f2bf(float f) {
  // round-to-nearest-even float -> bf16
  unsigned u = __builtin_bit_cast(unsigned, f);
  unsigned r = (u + 0x7FFFu + ((u >> 16) & 1u)) >> 16;
  return __builtin_bit_cast(bf16, (unsigned short)r);
}

union Frag { uint4 q[2]; v16bf v; };
union Pk4  { bf16 h[4]; uint2 u; };

// ---- decode: W[o][k] = codebook[o, indices[o*4096 + k]]  (fp32 -> bf16) ----
__global__ void __launch_bounds__(256)
decode_weights_kernel(const float* __restrict__ cb,
                      const long long* __restrict__ idx,
                      bf16* __restrict__ wb) {
  size_t base = ((size_t)blockIdx.x * 256 + threadIdx.x) * 4;   // 4 elems/thread
  const float* cbr = cb + ((base >> 12) << 8);                  // row o = base/4096, *256
  const long long* ip = idx + base;
  long long i0 = ip[0], i1 = ip[1], i2 = ip[2], i3 = ip[3];
  Pk4 o;
  o.h[0] = f2bf(cbr[(unsigned)i0 & 255u]);
  o.h[1] = f2bf(cbr[(unsigned)i1 & 255u]);
  o.h[2] = f2bf(cbr[(unsigned)i2 & 255u]);
  o.h[3] = f2bf(cbr[(unsigned)i3 & 255u]);
  *reinterpret_cast<uint2*>(wb + base) = o.u;
}

// ---- x fp32 -> bf16 ----
__global__ void __launch_bounds__(256)
convert_x_kernel(const float* __restrict__ x, bf16* __restrict__ xb) {
  size_t base = ((size_t)blockIdx.x * 256 + threadIdx.x) * 4;
  float4 f = *reinterpret_cast<const float4*>(x + base);
  Pk4 o;
  o.h[0] = f2bf(f.x); o.h[1] = f2bf(f.y); o.h[2] = f2bf(f.z); o.h[3] = f2bf(f.w);
  *reinterpret_cast<uint2*>(xb + base) = o.u;
}

// ---- bf16 WMMA GEMM: out[m][n] = sum_k xb[m][k]*wb[n][k] + bias[n] ----
// Block tile 128(M) x 256(N), 8 waves in 2x4, wave tile 64x64 = 16 WMMA / K-step.
__global__ void __launch_bounds__(256)
gemm_wmma_kernel(const bf16* __restrict__ xb, const bf16* __restrict__ wb,
                 const float* __restrict__ bias, float* __restrict__ out) {
  __shared__ bf16 sA[2][128 * LDSS];   // 2 x 10 KB
  __shared__ bf16 sB[2][256 * LDSS];   // 2 x 20 KB

  const int t    = threadIdx.x;
  const int lane = t & 31;
  const int wave = t >> 5;
  const int wm   = wave >> 2;          // 0..1 : 64 M-rows per wave
  const int wn   = wave & 3;           // 0..3 : 64 N-cols per wave
  const int half = lane >> 4;          // 0..1
  const int l16  = lane & 15;

  const int m0 = blockIdx.y * 128;
  const int n0 = blockIdx.x * 256;

  // global->LDS staging: per K-step each thread moves 2x16B (A) + 4x16B (B)
  const int lrow = t >> 2;             // 0..63
  const int lcol = (t & 3) * 8;        // 0,8,16,24 (bf16 units)

  const bf16* ga = xb + (size_t)(m0 + lrow) * K_DIM + lcol;
  const bf16* gb = wb + (size_t)(n0 + lrow) * K_DIM + lcol;

  v8f acc[4][4] = {};

  // prologue: stage K-tile 0
  {
    uint4 ra0 = *(const uint4*)(ga);
    uint4 ra1 = *(const uint4*)(ga + (size_t)64 * K_DIM);
    uint4 rb0 = *(const uint4*)(gb);
    uint4 rb1 = *(const uint4*)(gb + (size_t)64 * K_DIM);
    uint4 rb2 = *(const uint4*)(gb + (size_t)128 * K_DIM);
    uint4 rb3 = *(const uint4*)(gb + (size_t)192 * K_DIM);
    *(uint4*)&sA[0][lrow * LDSS + lcol]         = ra0;
    *(uint4*)&sA[0][(lrow + 64) * LDSS + lcol]  = ra1;
    *(uint4*)&sB[0][lrow * LDSS + lcol]         = rb0;
    *(uint4*)&sB[0][(lrow + 64) * LDSS + lcol]  = rb1;
    *(uint4*)&sB[0][(lrow + 128) * LDSS + lcol] = rb2;
    *(uint4*)&sB[0][(lrow + 192) * LDSS + lcol] = rb3;
  }
  ga += 32; gb += 32;
  __syncthreads();

  const int KT = K_DIM / 32;
  for (int kt = 0; kt < KT; ++kt) {
    const int  cur  = kt & 1;
    const bool more = (kt + 1) < KT;

    uint4 ra0, ra1, rb0, rb1, rb2, rb3;
    if (more) {  // issue next tile's global loads early (latency hidden by 16 WMMA)
      ra0 = *(const uint4*)(ga);
      ra1 = *(const uint4*)(ga + (size_t)64 * K_DIM);
      rb0 = *(const uint4*)(gb);
      rb1 = *(const uint4*)(gb + (size_t)64 * K_DIM);
      rb2 = *(const uint4*)(gb + (size_t)128 * K_DIM);
      rb3 = *(const uint4*)(gb + (size_t)192 * K_DIM);
      ga += 32; gb += 32;
    }

    // B fragments: lane l16 = column N, half selects K 0..15 / 16..31
    Frag fbj[4];
#pragma unroll
    for (int j = 0; j < 4; ++j) {
      const bf16* p = &sB[cur][(wn * 64 + j * 16 + l16) * LDSS + half * 16];
      fbj[j].q[0] = *(const uint4*)(p);
      fbj[j].q[1] = *(const uint4*)(p + 8);
    }

#pragma unroll
    for (int i = 0; i < 4; ++i) {
      // A fragment: lane l16 = row M; half selects K 0..15 / 16..31
      Frag fa;
      const bf16* p = &sA[cur][(wm * 64 + i * 16 + l16) * LDSS + half * 16];
      fa.q[0] = *(const uint4*)(p);
      fa.q[1] = *(const uint4*)(p + 8);
#pragma unroll
      for (int j = 0; j < 4; ++j) {
        acc[i][j] = __builtin_amdgcn_wmma_f32_16x16x32_bf16(
            false, fa.v, false, fbj[j].v, (short)0, acc[i][j], false, false);
      }
    }

    if (more) {  // fill the other buffer; prior-iter barrier made it free
      const int nxt = cur ^ 1;
      *(uint4*)&sA[nxt][lrow * LDSS + lcol]         = ra0;
      *(uint4*)&sA[nxt][(lrow + 64) * LDSS + lcol]  = ra1;
      *(uint4*)&sB[nxt][lrow * LDSS + lcol]         = rb0;
      *(uint4*)&sB[nxt][(lrow + 64) * LDSS + lcol]  = rb1;
      *(uint4*)&sB[nxt][(lrow + 128) * LDSS + lcol] = rb2;
      *(uint4*)&sB[nxt][(lrow + 192) * LDSS + lcol] = rb3;
    }
    __syncthreads();
  }

  // epilogue: C/D layout -> VGPR r holds M = r + half*8, lane l16 = N
#pragma unroll
  for (int i = 0; i < 4; ++i) {
#pragma unroll
    for (int j = 0; j < 4; ++j) {
      const int   n  = n0 + wn * 64 + j * 16 + l16;
      const float bj = bias[n];
#pragma unroll
      for (int r = 0; r < 8; ++r) {
        const int m = m0 + wm * 64 + i * 16 + half * 8 + r;
        out[(size_t)m * N_DIM + n] = acc[i][j][r] + bj;
      }
    }
  }
}

extern "C" void kernel_launch(void* const* d_in, const int* in_sizes, int n_in,
                              void* d_out, int out_size, void* d_ws, size_t ws_size,
                              hipStream_t stream) {
  const float*     x    = (const float*)d_in[0];
  const float*     cb   = (const float*)d_in[1];
  const float*     bias = (const float*)d_in[2];
  const long long* idx  = (const long long*)d_in[3];
  float* out = (float*)d_out;

  bf16* xb = (bf16*)d_ws;                          // 8192*4096 bf16 = 64 MB
  bf16* wb = xb + (size_t)M_DIM * K_DIM;           // 4096*4096 bf16 = 32 MB

  convert_x_kernel<<<((size_t)M_DIM * K_DIM) / (256 * 4), 256, 0, stream>>>(x, xb);
  decode_weights_kernel<<<((size_t)N_DIM * K_DIM) / (256 * 4), 256, 0, stream>>>(cb, idx, wb);

  dim3 grid(N_DIM / 256, M_DIM / 128);             // 16 x 64 = 1024 blocks
  gemm_wmma_kernel<<<grid, 256, 0, stream>>>(xb, wb, bias, out);
}
